// HebbHardConv2d_35502199669344
// MI455X (gfx1250) — matrix operand order repaired and checked by reading, not compiled, not run on hardware
//
#include <hip/hip_runtime.h>
#include <hip/hip_bf16.h>
#include <math.h>

// ---------------------------------------------------------------------------
// HebbHardConv2d forward on gfx1250 (CDNA5, wave32, WMMA + async global->LDS)
//   pre_x = conv2d(x, sign(w)*|w|^2)   [implicit GEMM, bf16 WMMA, f32 accum]
//   wta   = one_hot(argmax_c(pre_x))
// ---------------------------------------------------------------------------

typedef __attribute__((ext_vector_type(16))) __bf16 bf16x16;
typedef __attribute__((ext_vector_type(8)))  float  v8f;

#define B_SZ   32
#define C_IN   96
#define HW_    56
#define C_OUT  384
#define HWSQ   (HW_ * HW_)        // 3136
#define NPOS   (B_SZ * HWSQ)      // 100352
#define KTOT   (C_IN * 9)         // 864
#define LDS_STRIDE 40             // 32 K-elems + 8 pad (bf16) -> 80B rows

#define WA_BYTES  ((size_t)C_OUT * KTOT * 2)        //   663,552
#define XN_BYTES  ((size_t)NPOS * C_IN * 2)         //19,267,584
#define NEED_WS   (WA_BYTES + XN_BYTES)

// ---- CDNA5 async global->LDS (ASYNCcnt-tracked), via inline asm ------------
typedef __attribute__((address_space(3))) char lds_char;

__device__ __forceinline__ void async_b128(const void* g, const void* l) {
  asm volatile("global_load_async_to_lds_b128 %0, %1, off"
               :: "v"((lds_char*)l), "v"(g)
               : "memory");
}
__device__ __forceinline__ void wait_async0() {
  asm volatile("s_wait_asynccnt 0x0" ::: "memory");
}

// ---------------------------------------------------------------------------
// Prep kernels: bf16-pack operands into GEMM-friendly contiguous layouts.
// ---------------------------------------------------------------------------
__global__ __launch_bounds__(256)
void prep_w(const float* __restrict__ w, __bf16* __restrict__ wA) {
  const int i = blockIdx.x * 256 + threadIdx.x;     // [oc][tap][cin]
  if (i >= C_OUT * KTOT) return;
  const int oc = i / KTOT, r = i % KTOT, t = r / C_IN, cin = r % C_IN;
  const float v = w[(size_t)(oc * C_IN + cin) * 9 + t];
  wA[i] = (__bf16)(v * __builtin_fabsf(v));         // sign(w)*|w|^2
}

__global__ __launch_bounds__(256)
void prep_x(const float* __restrict__ x, __bf16* __restrict__ xn) {
  const size_t i = (size_t)blockIdx.x * 256 + threadIdx.x;  // [pos][cin]
  if (i >= (size_t)NPOS * C_IN) return;
  const int c = (int)(i % C_IN);
  const size_t pos = i / C_IN;
  const int b = (int)(pos / HWSQ), hw = (int)(pos % HWSQ);
  xn[i] = (__bf16)x[((size_t)b * C_IN + c) * HWSQ + hw];
}

// ---------------------------------------------------------------------------
// Fast path: double-buffered async-staged implicit GEMM.
// ---------------------------------------------------------------------------
__global__ __launch_bounds__(256)
void hebb_conv_wmma_async(const __bf16* __restrict__ wA,
                          const __bf16* __restrict__ xn,
                          float* __restrict__ pre) {
  __shared__ __bf16 ldsA[2][128 * LDS_STRIDE];
  __shared__ __bf16 ldsB[2][128 * LDS_STRIDE];

  const int tid  = threadIdx.x;
  const int lane = tid & 31;
  const int wid  = tid >> 5;
  const int wm   = wid >> 1;       // M offset wm*32
  const int wn   = wid & 1;        // N offset wn*64
  const int l15  = lane & 15;
  const int half = lane >> 4;

  const int nBase  = blockIdx.x * 128;
  const int ocBase = blockIdx.y * 128;

  // staging: each thread owns 16B segment s4 of rows r0 and r0+64 (A and B)
  const int s4 = tid & 3;
  const int r0 = tid >> 2;                      // 0..63
  const int ng0 = nBase + r0;
  const int ng1 = nBase + r0 + 64;
  const int b0 = ng0 / HWSQ, hw0 = ng0 % HWSQ, h0 = hw0 / HW_, w0 = hw0 % HW_;
  const int b1 = ng1 / HWSQ, hw1 = ng1 % HWSQ, h1 = hw1 / HW_, w1 = hw1 % HW_;

  const __bf16* gA0 = wA + (size_t)(ocBase + r0)      * KTOT + s4 * 8;
  const __bf16* gA1 = wA + (size_t)(ocBase + r0 + 64) * KTOT + s4 * 8;

  v8f acc[2][4] = {};
  int buf = 0;

  auto stage = [&](int bsel, int kc) {
    const int t  = kc / 3;
    const int c0 = (kc % 3) * 32;
    const int dy = t / 3 - 1, dx = t % 3 - 1;
    const int koff = t * C_IN + c0;
    // A tile: 128 x 64B rows, fully contiguous in wA
    async_b128(gA0 + koff, &ldsA[bsel][r0        * LDS_STRIDE + s4 * 8]);
    async_b128(gA1 + koff, &ldsA[bsel][(r0 + 64) * LDS_STRIDE + s4 * 8]);
    // B tile: im2col rows from NHWC bf16 x; zero-fill conv borders via DS
    const float4 z4 = {0.f, 0.f, 0.f, 0.f};
    {
      const int hy = h0 + dy, wx = w0 + dx;
      __bf16* l = &ldsB[bsel][r0 * LDS_STRIDE + s4 * 8];
      if ((unsigned)hy < (unsigned)HW_ && (unsigned)wx < (unsigned)HW_)
        async_b128(xn + ((size_t)b0 * HWSQ + hy * HW_ + wx) * C_IN + c0 + s4 * 8, l);
      else
        *(float4*)l = z4;
    }
    {
      const int hy = h1 + dy, wx = w1 + dx;
      __bf16* l = &ldsB[bsel][(r0 + 64) * LDS_STRIDE + s4 * 8];
      if ((unsigned)hy < (unsigned)HW_ && (unsigned)wx < (unsigned)HW_)
        async_b128(xn + ((size_t)b1 * HWSQ + hy * HW_ + wx) * C_IN + c0 + s4 * 8, l);
      else
        *(float4*)l = z4;
    }
  };

  stage(0, 0);
  wait_async0();
  __syncthreads();

  for (int kc = 0; kc < 27; ++kc) {
    if (kc + 1 < 27) stage(buf ^ 1, kc + 1);   // prefetch next chunk (async DMA)

    // fragment loads per ISA 16-bit A(16x32)/B(32x16) lane layouts
    union FragU { bf16x16 v; float4 f[2]; };
    FragU a[2], b[4];
    #pragma unroll
    for (int mt = 0; mt < 2; ++mt) {
      const int row = wm * 32 + mt * 16 + l15;
      const __bf16* p = &ldsA[buf][row * LDS_STRIDE];
      a[mt].f[0] = *(const float4*)(p + 8 * half);
      a[mt].f[1] = *(const float4*)(p + 16 + 8 * half);
    }
    #pragma unroll
    for (int nt = 0; nt < 4; ++nt) {
      const int row = wn * 64 + nt * 16 + l15;
      const __bf16* p = &ldsB[buf][row * LDS_STRIDE + 8 * half];
      b[nt].f[0] = *(const float4*)(p);
      b[nt].f[1] = *(const float4*)(p + 16);
    }
    #pragma unroll
    for (int mt = 0; mt < 2; ++mt)
      #pragma unroll
      for (int nt = 0; nt < 4; ++nt)
        acc[mt][nt] = __builtin_amdgcn_wmma_f32_16x16x32_bf16(
            false, a[mt].v, false, b[nt].v, (short)0, acc[mt][nt], false, false);

    wait_async0();      // this wave's DMAs for chunk kc+1 landed
    __syncthreads();    // everyone done reading buf / writing buf^1
    buf ^= 1;
  }

  // epilogue: C/D layout VGPR r -> M = r + 8*half, N = lane&15
  #pragma unroll
  for (int mt = 0; mt < 2; ++mt) {
    #pragma unroll
    for (int nt = 0; nt < 4; ++nt) {
      const int ocb = ocBase + wm * 32 + mt * 16 + 8 * half;
      const int ng  = nBase + wn * 64 + nt * 16 + l15;
      const int bb  = ng / HWSQ;
      const int hw  = ng % HWSQ;
      float* o = pre + (size_t)bb * C_OUT * HWSQ + hw;
      #pragma unroll
      for (int r = 0; r < 8; ++r)
        o[(size_t)(ocb + r) * HWSQ] = acc[mt][nt][r];
    }
  }
}

// ---------------------------------------------------------------------------
// Fallback path (no workspace): in-kernel convert + DS staging (round-1).
// ---------------------------------------------------------------------------
__global__ __launch_bounds__(256)
void hebb_conv_wmma_fb(const float* __restrict__ x, const float* __restrict__ weight,
                       float* __restrict__ pre) {
  __shared__ __bf16 ldsA[128 * LDS_STRIDE];
  __shared__ __bf16 ldsB[128 * LDS_STRIDE];

  const int tid  = threadIdx.x;
  const int lane = tid & 31;
  const int wid  = tid >> 5;
  const int wm   = wid >> 1;
  const int wn   = wid & 1;
  const int l15  = lane & 15;
  const int half = lane >> 4;

  const int nBase  = blockIdx.x * 128;
  const int ocBase = blockIdx.y * 128;

  const int kkA  = tid & 31;
  const int mA0  = tid >> 5;
  const int nB   = tid & 127;
  const int kkB0 = tid >> 7;

  const int ngF = nBase + nB;
  const int bF  = ngF / HWSQ;
  const int hwF = ngF % HWSQ;
  const int hF  = hwF / HW_;
  const int wF  = hwF % HW_;
  const float* xb = x + (size_t)bF * C_IN * HWSQ;

  v8f acc[2][4] = {};

  for (int kc = 0; kc < 27; ++kc) {
    const int t  = kc / 3;
    const int c0 = (kc % 3) * 32;
    const int dy = t / 3, dx = t % 3;

    __syncthreads();
    {
      const int cin = c0 + kkA;
      #pragma unroll
      for (int it = 0; it < 16; ++it) {
        const int m = it * 8 + mA0;
        const float wv = weight[(size_t)((ocBase + m) * C_IN + cin) * 9 + t];
        ldsA[m * LDS_STRIDE + kkA] = (__bf16)(wv * __builtin_fabsf(wv));
      }
    }
    {
      const int hy = hF + dy - 1;
      const int wx = wF + dx - 1;
      const bool inb = ((unsigned)hy < (unsigned)HW_) && ((unsigned)wx < (unsigned)HW_);
      const size_t sp = (size_t)hy * HW_ + wx;
      #pragma unroll
      for (int it = 0; it < 16; ++it) {
        const int kk = it * 2 + kkB0;
        float xv = 0.0f;
        if (inb) xv = xb[(size_t)(c0 + kk) * HWSQ + sp];
        ldsB[nB * LDS_STRIDE + kk] = (__bf16)xv;
      }
    }
    __syncthreads();

    union FragU { bf16x16 v; float4 f[2]; };
    FragU a[2], b[4];
    #pragma unroll
    for (int mt = 0; mt < 2; ++mt) {
      const int row = wm * 32 + mt * 16 + l15;
      const __bf16* p = &ldsA[row * LDS_STRIDE];
      a[mt].f[0] = *(const float4*)(p + 8 * half);
      a[mt].f[1] = *(const float4*)(p + 16 + 8 * half);
    }
    #pragma unroll
    for (int nt = 0; nt < 4; ++nt) {
      const int row = wn * 64 + nt * 16 + l15;
      const __bf16* p = &ldsB[row * LDS_STRIDE + 8 * half];
      b[nt].f[0] = *(const float4*)(p);
      b[nt].f[1] = *(const float4*)(p + 16);
    }
    #pragma unroll
    for (int mt = 0; mt < 2; ++mt)
      #pragma unroll
      for (int nt = 0; nt < 4; ++nt)
        acc[mt][nt] = __builtin_amdgcn_wmma_f32_16x16x32_bf16(
            false, a[mt].v, false, b[nt].v, (short)0, acc[mt][nt], false, false);
  }

  #pragma unroll
  for (int mt = 0; mt < 2; ++mt) {
    #pragma unroll
    for (int nt = 0; nt < 4; ++nt) {
      const int ocb = ocBase + wm * 32 + mt * 16 + 8 * half;
      const int ng  = nBase + wn * 64 + nt * 16 + l15;
      const int bb  = ng / HWSQ;
      const int hw  = ng % HWSQ;
      float* o = pre + (size_t)bb * C_OUT * HWSQ + hw;
      #pragma unroll
      for (int r = 0; r < 8; ++r)
        o[(size_t)(ocb + r) * HWSQ] = acc[mt][nt][r];
    }
  }
}

// Winner-take-all: channel argmax (first-max tie-break, like jnp.argmax) + one-hot.
__global__ __launch_bounds__(256)
void hebb_wta(const float* __restrict__ pre, float* __restrict__ wta) {
  const int pos = blockIdx.x * 256 + threadIdx.x;
  if (pos >= NPOS) return;
  const int bb = pos / HWSQ;
  const int hw = pos % HWSQ;
  const float* p = pre + (size_t)bb * C_OUT * HWSQ + hw;
  int best = 0;
  float bv = p[0];
  for (int c = 1; c < C_OUT; ++c) {
    const float v = p[(size_t)c * HWSQ];
    if (v > bv) { bv = v; best = c; }
  }
  float* o = wta + (size_t)bb * C_OUT * HWSQ + hw;
  for (int c = 0; c < C_OUT; ++c)
    o[(size_t)c * HWSQ] = (c == best) ? 1.0f : 0.0f;
}

extern "C" void kernel_launch(void* const* d_in, const int* in_sizes, int n_in,
                              void* d_out, int out_size, void* d_ws, size_t ws_size,
                              hipStream_t stream) {
  const float* x = (const float*)d_in[0];
  const float* w = (const float*)d_in[1];
  float* pre = (float*)d_out;
  float* wta = pre + (size_t)B_SZ * C_OUT * HWSQ;

  dim3 grid(NPOS / 128, C_OUT / 128, 1);   // 784 x 3

  if (ws_size >= NEED_WS) {
    __bf16* wA = (__bf16*)d_ws;
    __bf16* xn = (__bf16*)((char*)d_ws + WA_BYTES);
    prep_w<<<(C_OUT * KTOT + 255) / 256, 256, 0, stream>>>(w, wA);
    prep_x<<<(int)(((size_t)NPOS * C_IN + 255) / 256), 256, 0, stream>>>(x, xn);
    hebb_conv_wmma_async<<<grid, 256, 0, stream>>>(wA, xn, pre);
  } else {
    hebb_conv_wmma_fb<<<grid, 256, 0, stream>>>(x, w, pre);
  }
  hebb_wta<<<(NPOS + 255) / 256, 256, 0, stream>>>(pre, wta);
}
